// LlamaAttention_34961033789605
// MI455X (gfx1250) — matrix operand order repaired and checked by reading, not compiled, not run on hardware
//
#include <hip/hip_runtime.h>
#include <hip/hip_bf16.h>

typedef __attribute__((ext_vector_type(16))) __bf16 v16bf;
typedef __attribute__((ext_vector_type(8)))  __bf16 v8bf;
typedef __attribute__((ext_vector_type(4)))  __bf16 v4bf;
typedef __attribute__((ext_vector_type(8)))  float  v8f;
typedef __attribute__((ext_vector_type(4)))  float  v4f;

constexpr int Bc = 2, Sc = 2048, Dc = 2048, Hc = 32, KVHc = 8, HDc = 64;
constexpr int Mrows = Bc * Sc;     // 4096
constexpr int NQ    = Hc * HDc;    // 2048
constexpr int NKV   = KVHc * HDc;  // 512

#define ASYNC_STAGE 1

// ---------------- WMMA helpers (CDNA5 16x16x32 bf16, f32 accum) ----------------

__device__ __forceinline__ v8f wmma_bf16(v16bf a, v16bf b, v8f c) {
  return __builtin_amdgcn_wmma_f32_16x16x32_bf16(false, a, false, b, (short)0, c,
                                                 false, false);
}

// A-matrix 16x32 (MxK), bf16, contiguous k order (same as load_b_bf16):
//   lanes 0-15 : M=lane,    K in {0..15}
//   lanes 16-31: M=lane-16, K in {16..31}
__device__ __forceinline__ v16bf load_a_bf16(const __bf16* base, int ld, int m0,
                                             int k0, int lane) {
  const int mr = lane & 15, hf = lane >> 4;
  const __bf16* p = base + (size_t)(m0 + mr) * ld + k0 + hf * 16;
  v8bf lo = *(const v8bf*)p;
  v8bf hi = *(const v8bf*)(p + 8);
  v16bf a;
#pragma unroll
  for (int i = 0; i < 8; ++i) { a[i] = lo[i]; a[8 + i] = hi[i]; }
  return a;
}

// B-matrix 32x16 (KxN), bf16: lanes 0-15 hold K=0..15 for N=lane,
// lanes 16-31 hold K=16..31 for N=lane-16. Source row-major (N x Kstride):
// B[k][n] = src[(n0+n)*ld + k0+k]  (16 contiguous elements per lane).
__device__ __forceinline__ v16bf load_b_bf16(const __bf16* src, int ld, int n0,
                                             int k0, int lane) {
  const int n = lane & 15, kh = (lane >> 4) * 16;
  return *(const v16bf*)(src + (size_t)(n0 + n) * ld + k0 + kh);
}

// C/D 16x16 f32: VGPR r, lanes 0-15 -> (M=r, N=lane); lanes 16-31 -> (M=r+8, ...)
__device__ __forceinline__ void store_c_f32(float* C, int ld, int m0, int n0,
                                            int lane, const v8f& acc) {
  const int n = n0 + (lane & 15);
  const int mb = m0 + (lane >> 4) * 8;
#pragma unroll
  for (int r = 0; r < 8; ++r) C[(size_t)(mb + r) * ld + n] = acc[r];
}

// ---------------- CDNA5 async global->LDS staging ------------------------------

__device__ __forceinline__ void async_copy16(const __bf16* g, __bf16* l) {
#if ASYNC_STAGE
  // VDST holds the LDS byte address (low 32 bits of the flat pointer),
  // VADDR the 64-bit global address.  Tracked with ASYNCcnt.
  unsigned loff = (unsigned)(size_t)(void*)l;
  asm volatile("global_load_async_to_lds_b128 %0, %1, off"
               :: "v"(loff), "v"(g) : "memory");
#else
  *(v8bf*)l = *(const v8bf*)g;
#endif
}

__device__ __forceinline__ void async_wait_prev4() {  // 4 next-tile ops in flight
#if ASYNC_STAGE
  asm volatile("s_wait_asynccnt 0x4" ::: "memory");
#endif
}
__device__ __forceinline__ void async_wait_prev8() {  // 8 next-tile ops in flight
#if ASYNC_STAGE
  asm volatile("s_wait_asynccnt 0x8" ::: "memory");
#endif
}
__device__ __forceinline__ void async_wait_all() {
#if ASYNC_STAGE
  asm volatile("s_wait_asynccnt 0x0" ::: "memory");
#endif
}

// ---------------- fp32 -> bf16 bulk convert ------------------------------------

__global__ __launch_bounds__(256) void cvt_bf16_kernel(const float* __restrict__ src,
                                                       __bf16* __restrict__ dst,
                                                       int n4) {
  const int i = blockIdx.x * blockDim.x + threadIdx.x;
  if (i >= n4) return;
  v4f s = *(const v4f*)(src + (size_t)i * 4);
  v4bf d;
#pragma unroll
  for (int k = 0; k < 4; ++k) d[k] = (__bf16)s[k];
  *(v4bf*)(dst + (size_t)i * 4) = d;
}

// ---------------- GEMM: C[M,N] = A[M,K] * W[N,K]^T  (all-bf16 operands) --------
// 128x128 block tile, 8 waves (4 down M x 2 across N), K-step 32,
// double-buffered async global->LDS staging, fragments served from LDS.

__global__ __launch_bounds__(256) void gemm_bf16_lds_kernel(
    const __bf16* __restrict__ A, const __bf16* __restrict__ W,
    float* __restrict__ C, int M, int N, int K) {
  constexpr int KS = 32;
  constexpr int LDT = 40;  // padded row stride: bank starts 20n mod 64, spacing 4
  __shared__ __bf16 shA[2][128 * LDT];
  __shared__ __bf16 shB[2][128 * LDT];

  const int tid = threadIdx.x;
  const int lane = tid & 31;
  const int wave = tid >> 5;
  const int bm0 = blockIdx.y * 128;
  const int bn0 = blockIdx.x * 128;
  const int wm = (wave >> 1) * 32;
  const int wn = (wave & 1) * 64;

  auto stage = [&](int t, int buf) {
    const int k0 = t * KS;
#pragma unroll
    for (int i = 0; i < 2; ++i) {
      const int c = tid + 256 * i;          // 512 16B-chunks per 128x32 tile
      const int row = c >> 2, col = (c & 3) * 8;
      async_copy16(A + (size_t)(bm0 + row) * K + k0 + col,
                   &shA[buf][row * LDT + col]);
      async_copy16(W + (size_t)(bn0 + row) * K + k0 + col,
                   &shB[buf][row * LDT + col]);
    }
  };

  v8f acc[2][4] = {};
  const int T = K / KS;

  stage(0, 0);
  for (int t = 0; t < T; ++t) {
    const int buf = t & 1;
    if (t + 1 < T) {
      stage(t + 1, buf ^ 1);  // overlap next tile with this tile's math
      async_wait_prev4();
    } else {
      async_wait_all();
    }
    __syncthreads();

    v16bf a0 = load_a_bf16(shA[buf], LDT, wm, 0, lane);
    v16bf a1 = load_a_bf16(shA[buf], LDT, wm + 16, 0, lane);
#pragma unroll
    for (int j = 0; j < 4; ++j) {
      v16bf bf = load_b_bf16(shB[buf], LDT, wn + 16 * j, 0, lane);
      acc[0][j] = wmma_bf16(a0, bf, acc[0][j]);
      acc[1][j] = wmma_bf16(a1, bf, acc[1][j]);
    }
    __syncthreads();
  }

#pragma unroll
  for (int i = 0; i < 2; ++i)
#pragma unroll
    for (int j = 0; j < 4; ++j)
      store_c_f32(C, N, bm0 + wm + 16 * i, bn0 + wn + 16 * j, lane, acc[i][j]);
}

// ---------------- RoPE + bf16 convert + head-major transpose -------------------
// Xf: (B*S, heads*64) fp32  ->  Xb: (B, heads, S, 64) bf16 with RoPE applied.
__global__ void rope_cvt_kernel(const float* __restrict__ Xf,
                                const float* __restrict__ cosp,
                                const float* __restrict__ sinp,
                                __bf16* __restrict__ Xb, int heads) {
  const int tid = blockIdx.x * blockDim.x + threadIdx.x;
  const int total = Bc * Sc * heads * 32;
  if (tid >= total) return;
  const int j = tid & 31;
  const int h = (tid >> 5) % heads;
  const int s = (tid / (32 * heads)) % Sc;
  const int b = tid / (32 * heads * Sc);

  const size_t m = (size_t)b * Sc + s;
  const int ldx = heads * HDc;
  const float q1 = Xf[m * ldx + h * HDc + j];
  const float q2 = Xf[m * ldx + h * HDc + j + 32];
  const float* cp = cosp + m * HDc;
  const float* sp = sinp + m * HDc;
  const float o1 = q1 * cp[j] - q2 * sp[j];
  const float o2 = q2 * cp[j + 32] + q1 * sp[j + 32];

  __bf16* dst = Xb + (((size_t)b * heads + h) * Sc + s) * HDc;
  dst[j] = (__bf16)o1;
  dst[j + 32] = (__bf16)o2;
}

// V: (B*S, KVH*64) fp32 -> Vt: (B, KVH, 64, S) bf16 (transposed for B-frags)
__global__ void v_cvt_kernel(const float* __restrict__ Vf, __bf16* __restrict__ Vt) {
  const int tid = blockIdx.x * blockDim.x + threadIdx.x;
  const int total = Bc * Sc * KVHc * HDc;
  if (tid >= total) return;
  const int j = tid & 63;
  const int h = (tid >> 6) % KVHc;
  const int s = (tid / (HDc * KVHc)) % Sc;
  const int b = tid / (HDc * KVHc * Sc);
  const float v = Vf[((size_t)b * Sc + s) * NKV + h * HDc + j];
  Vt[(((size_t)b * KVHc + h) * HDc + j) * Sc + s] = (__bf16)v;
}

// ---------------- Flash attention (online softmax, causal, GQA) ---------------
// grid: (S/64, H, B), block: 128 (4 waves, 16 query rows each).
// K/V tiles are staged once per block via async global->LDS (double-buffered)
// and shared by all 4 waves: 4x less L2 traffic than per-wave fragment loads.
__global__ __launch_bounds__(128) void flash_attn_kernel(
    const __bf16* __restrict__ Qb,   // (B, H,   S, 64)
    const __bf16* __restrict__ Kb,   // (B, KVH, S, 64)
    const __bf16* __restrict__ Vt,   // (B, KVH, 64, S)
    __bf16* __restrict__ Ab) {       // (B, S, H*64)
  constexpr int LDT = 72;  // padded stride: bank starts 36n mod 64, spacing 4
  __shared__ __bf16 shK[2][64 * LDT];  // (key, hd) tile
  __shared__ __bf16 shV[2][64 * LDT];  // (hd, key) tile
  __shared__ __bf16 Pl[4][16][64];     // per-wave P staging (C -> A layout)

  const int tid = threadIdx.x;
  const int lane = tid & 31;
  const int wave = tid >> 5;  // 0..3
  const int qb = blockIdx.x;
  const int h = blockIdx.y;
  const int b = blockIdx.z;
  const int kvh = h / (Hc / KVHc);
  const int m0 = qb * 64 + wave * 16;
  const int nl = lane & 15, hf = lane >> 4;

  const __bf16* Qh = Qb + ((size_t)(b * Hc + h) * Sc) * HDc;
  const __bf16* Kh = Kb + ((size_t)(b * KVHc + kvh) * Sc) * HDc;
  const __bf16* Vh = Vt + ((size_t)(b * KVHc + kvh) * HDc) * Sc;

  const v16bf qa0 = load_a_bf16(Qh, HDc, m0, 0, lane);
  const v16bf qa1 = load_a_bf16(Qh, HDc, m0, 32, lane);

  // cooperative stage of one 64-key K tile + V tile (8 KB each, 512 chunks)
  auto stageKV = [&](int ti, int buf) {
    const int kb = ti * 64;
#pragma unroll
    for (int i = 0; i < 4; ++i) {
      const int c = tid + 128 * i;
      const int row = c >> 3, col = (c & 7) * 8;
      async_copy16(Kh + (size_t)(kb + row) * HDc + col,
                   &shK[buf][row * LDT + col]);
      async_copy16(Vh + (size_t)row * Sc + kb + col,
                   &shV[buf][row * LDT + col]);
    }
  };

  float rmax[8], rsum[8];
#pragma unroll
  for (int r = 0; r < 8; ++r) { rmax[r] = -1e30f; rsum[r] = 0.0f; }
  v8f o[4] = {};

  const int T = qb + 1;  // causal: 64-key tiles up to the diagonal
  stageKV(0, 0);
  for (int ti = 0; ti < T; ++ti) {
    const int kb = ti * 64;
    const int buf = ti & 1;
    if (ti + 1 < T) {
      stageKV(ti + 1, buf ^ 1);  // next tile DMA overlaps this tile's math
      async_wait_prev8();
    } else {
      async_wait_all();
    }
    __syncthreads();

    // ---- S = Q * K^T from shared K tile ----
    v8f sc[4];
#pragma unroll
    for (int j = 0; j < 4; ++j) {
      v16bf kf0 = load_b_bf16(shK[buf], LDT, 16 * j, 0, lane);
      v16bf kf1 = load_b_bf16(shK[buf], LDT, 16 * j, 32, lane);
      v8f z = {};
      z = wmma_bf16(qa0, kf0, z);
      z = wmma_bf16(qa1, kf1, z);
      sc[j] = z;
    }
    // ---- scale, causal mask, online softmax ----
#pragma unroll
    for (int r = 0; r < 8; ++r) {
      const int qi = m0 + hf * 8 + r;
      float rowm = -3.0e38f;
#pragma unroll
      for (int j = 0; j < 4; ++j) {
        const int kcol = kb + 16 * j + nl;
        float v = sc[j][r] * 0.125f;  // HD^-0.5
        v = (kcol <= qi) ? v : -3.0e38f;
        sc[j][r] = v;
        rowm = fmaxf(rowm, v);
      }
      rowm = fmaxf(rowm, __shfl_xor(rowm, 1));
      rowm = fmaxf(rowm, __shfl_xor(rowm, 2));
      rowm = fmaxf(rowm, __shfl_xor(rowm, 4));
      rowm = fmaxf(rowm, __shfl_xor(rowm, 8));
      const float nmax = fmaxf(rmax[r], rowm);
      const float corr = __expf(rmax[r] - nmax);
      rsum[r] *= corr;
#pragma unroll
      for (int j = 0; j < 4; ++j) o[j][r] *= corr;
      float lsum = 0.0f;
#pragma unroll
      for (int j = 0; j < 4; ++j) {
        const float e = __expf(sc[j][r] - nmax);
        sc[j][r] = e;
        lsum += e;
      }
      lsum += __shfl_xor(lsum, 1);
      lsum += __shfl_xor(lsum, 2);
      lsum += __shfl_xor(lsum, 4);
      lsum += __shfl_xor(lsum, 8);
      rsum[r] += lsum;
      rmax[r] = nmax;
    }
    // ---- P: C-layout -> per-wave LDS tile, reload as A fragments ----
#pragma unroll
    for (int j = 0; j < 4; ++j)
#pragma unroll
      for (int r = 0; r < 8; ++r)
        Pl[wave][hf * 8 + r][16 * j + nl] = (__bf16)sc[j][r];

    const v16bf pa0 = load_a_bf16(&Pl[wave][0][0], 64, 0, 0, lane);
    const v16bf pa1 = load_a_bf16(&Pl[wave][0][0], 64, 0, 32, lane);

    // ---- O += P * V from shared V tile ----
#pragma unroll
    for (int j = 0; j < 4; ++j) {
      v16bf vb0 = load_b_bf16(shV[buf], LDT, 16 * j, 0, lane);
      v16bf vb1 = load_b_bf16(shV[buf], LDT, 16 * j, 32, lane);
      o[j] = wmma_bf16(pa0, vb0, o[j]);
      o[j] = wmma_bf16(pa1, vb1, o[j]);
    }
    __syncthreads();
  }

#pragma unroll
  for (int j = 0; j < 4; ++j)
#pragma unroll
    for (int r = 0; r < 8; ++r) {
      const int qi = m0 + hf * 8 + r;
      Ab[((size_t)b * Sc + qi) * NQ + h * HDc + 16 * j + nl] =
          (__bf16)(o[j][r] / rsum[r]);
    }
}

// ---------------- Host-side launcher ------------------------------------------

extern "C" void kernel_launch(void* const* d_in, const int* in_sizes, int n_in,
                              void* d_out, int out_size, void* d_ws, size_t ws_size,
                              hipStream_t stream) {
  (void)in_sizes; (void)n_in; (void)out_size; (void)ws_size;
  const float* x    = (const float*)d_in[0];
  const float* cosp = (const float*)d_in[1];
  const float* sinp = (const float*)d_in[2];
  const float* Wq   = (const float*)d_in[3];
  const float* Wk   = (const float*)d_in[4];
  const float* Wv   = (const float*)d_in[5];
  const float* Wo   = (const float*)d_in[6];
  float* out = (float*)d_out;

  char* ws = (char*)d_ws;
  size_t off = 0;
  auto take = [&](size_t bytes) {
    char* p = ws + off;
    off += (bytes + 255) & ~(size_t)255;
    return p;
  };
  __bf16* xb  = (__bf16*)take((size_t)Mrows * Dc  * sizeof(__bf16));
  __bf16* Wqb = (__bf16*)take((size_t)NQ * Dc     * sizeof(__bf16));
  __bf16* Wkb = (__bf16*)take((size_t)NKV * Dc    * sizeof(__bf16));
  __bf16* Wvb = (__bf16*)take((size_t)NKV * Dc    * sizeof(__bf16));
  __bf16* Wob = (__bf16*)take((size_t)Dc * NQ     * sizeof(__bf16));
  float*  Qf  = (float*)take((size_t)Mrows * NQ   * sizeof(float));
  float*  Kf  = (float*)take((size_t)Mrows * NKV  * sizeof(float));
  float*  Vf  = (float*)take((size_t)Mrows * NKV  * sizeof(float));
  __bf16* Qb  = (__bf16*)take((size_t)Mrows * NQ  * sizeof(__bf16));
  __bf16* Kb  = (__bf16*)take((size_t)Mrows * NKV * sizeof(__bf16));
  __bf16* Vt  = (__bf16*)take((size_t)Mrows * NKV * sizeof(__bf16));
  __bf16* Ab  = (__bf16*)take((size_t)Mrows * NQ  * sizeof(__bf16));

  const dim3 blk(256);
  auto cvt = [&](const float* s, __bf16* d, size_t n) {
    int n4 = (int)(n / 4);
    cvt_bf16_kernel<<<(n4 + 255) / 256, blk, 0, stream>>>(s, d, n4);
  };
  // 0: one-shot fp32 -> bf16 of all GEMM operands (strips cvts from hot loops)
  cvt(x,  xb,  (size_t)Mrows * Dc);
  cvt(Wq, Wqb, (size_t)NQ * Dc);
  cvt(Wk, Wkb, (size_t)NKV * Dc);
  cvt(Wv, Wvb, (size_t)NKV * Dc);
  cvt(Wo, Wob, (size_t)Dc * NQ);

  // 1-3: QKV projections (bf16 WMMA, f32 accum, async double-buffered LDS)
  gemm_bf16_lds_kernel<<<dim3(NQ / 128, Mrows / 128), blk, 0, stream>>>(
      xb, Wqb, Qf, Mrows, NQ, Dc);
  gemm_bf16_lds_kernel<<<dim3(NKV / 128, Mrows / 128), blk, 0, stream>>>(
      xb, Wkb, Kf, Mrows, NKV, Dc);
  gemm_bf16_lds_kernel<<<dim3(NKV / 128, Mrows / 128), blk, 0, stream>>>(
      xb, Wvb, Vf, Mrows, NKV, Dc);

  // 4-6: RoPE + bf16 conversion + layout changes
  {
    int tq = Bc * Sc * Hc * 32;
    rope_cvt_kernel<<<(tq + 255) / 256, blk, 0, stream>>>(Qf, cosp, sinp, Qb, Hc);
    int tk = Bc * Sc * KVHc * 32;
    rope_cvt_kernel<<<(tk + 255) / 256, blk, 0, stream>>>(Kf, cosp, sinp, Kb, KVHc);
    int tv = Bc * Sc * KVHc * HDc;
    v_cvt_kernel<<<(tv + 255) / 256, blk, 0, stream>>>(Vf, Vt);
  }

  // 7: causal flash attention with GQA (async-staged shared K/V tiles)
  flash_attn_kernel<<<dim3(Sc / 64, Hc, Bc), dim3(128), 0, stream>>>(Qb, Kb, Vt, Ab);

  // 8: output projection (bf16 x bf16 -> fp32 out)
  gemm_bf16_lds_kernel<<<dim3(Dc / 128, Mrows / 128), blk, 0, stream>>>(
      Ab, Wob, out, Mrows, Dc, NQ);
}
